// MoELayerWithSkip_5686536700358
// MI455X (gfx1250) — matrix operand order repaired and checked by reading, not compiled, not run on hardware
//
#include <hip/hip_runtime.h>
#include <hip/hip_bf16.h>

// ---------------------------------------------------------------------------
// MoE layer with skip for MI455X (gfx1250): bf16 WMMA + TDM path.
// N=4096 tokens, D=768, H=3072, E=4 experts, K=2 (top-k).
// ---------------------------------------------------------------------------

#define NTOK 4096
#define DDIM 768
#define HDIM 3072
#define NEXP 4

typedef __bf16 bf16_t;
typedef __attribute__((ext_vector_type(16))) __bf16 v16bf;
typedef __attribute__((ext_vector_type(8)))  float  v8f;
typedef __attribute__((ext_vector_type(4)))  int    int4v;
typedef __attribute__((ext_vector_type(4)))  unsigned u32x4;
typedef __attribute__((ext_vector_type(8)))  int    i32x8;
typedef __attribute__((ext_vector_type(4)))  int    i32x4;

#if defined(__has_builtin)
#if __has_builtin(__builtin_amdgcn_tensor_load_to_lds) && \
    __has_builtin(__builtin_amdgcn_s_wait_tensorcnt)
#define USE_TDM 1
#endif
#endif

union FragBits { v16bf v; int4v q[2]; };

__device__ __forceinline__ v8f wmma_bf16(v16bf a, v16bf b, v8f c) {
  // D = A(16x32 bf16) * B(32x16 bf16) + C(16x16 f32)
  return __builtin_amdgcn_wmma_f32_16x16x32_bf16(
      /*neg_a=*/false, a, /*neg_b=*/false, b,
      /*c_mod=*/(short)0, c, /*reuse_a=*/false, /*reuse_b=*/false);
}

// A fragment from chunk-major LDS xn tile: layout [96 chunks][16 rows][8 bf16].
//   lanes 0-15 (hi=0): K = kd+0..7 (chunk kd/8)   and kd+8..15 (chunk kd/8+1)
//   lanes 16-31(hi=1): K = kd+16..23 (chunk kd/8+2) and kd+24..31 (chunk kd/8+3)
__device__ __forceinline__ v16bf load_frag_a_xn(const bf16_t* xs, int m, int kd, int hi) {
  FragBits f;
  const int c0 = (kd >> 3) + 2 * hi;
  f.q[0] = *(const int4v*)(xs + (c0 * 16 + m) * 8);
  f.q[1] = *(const int4v*)(xs + ((c0 + 1) * 16 + m) * 8);
  return f.v;
}

// A fragment from a row-major (padded-stride) LDS row.
__device__ __forceinline__ v16bf load_frag_a_row(const bf16_t* row, int kd, int hi) {
  FragBits f;
  const bf16_t* p0 = row + kd + (hi ? 16 : 0);
  f.q[0] = *(const int4v*)(p0);
  f.q[1] = *(const int4v*)(p0 + 8);
  return f.v;
}

// ---------------------------------------------------------------------------
// Kernel 0: convert weights to bf16 with transpose so WMMA B-fragments are
// contiguous 32-byte per-lane loads.
//   W1t[e][h][d] = bf16( W1[e][d][h] )   (GEMM1 B: K=d contiguous per h-row)
//   W2t[e][d][h] = bf16( W2[e][h][d] )   (GEMM2 B: K=h contiguous per d-row)
// ---------------------------------------------------------------------------
__global__ void prep_weights(const float* __restrict__ W1,
                             const float* __restrict__ W2,
                             bf16_t* __restrict__ W1t,
                             bf16_t* __restrict__ W2t) {
  const int total = NEXP * HDIM * DDIM;
  for (int idx = blockIdx.x * blockDim.x + threadIdx.x; idx < total;
       idx += gridDim.x * blockDim.x) {
    int e = idx / (HDIM * DDIM);
    int r = idx - e * (HDIM * DDIM);
    int h1 = r / DDIM, d1 = r - h1 * DDIM;
    W1t[idx] = (bf16_t)W1[((size_t)e * DDIM + d1) * HDIM + h1];
    int d2 = r / HDIM, h2 = r - d2 * HDIM;
    W2t[idx] = (bf16_t)W2[((size_t)e * HDIM + h2) * DDIM + d2];
  }
}

// ---------------------------------------------------------------------------
// Kernel 1: LayerNorm + router (one wave32 per token).
// Writes xn (bf16 [N,D]) and dense combine weights wcomb (f32 [N,E]).
// ---------------------------------------------------------------------------
__global__ __launch_bounds__(256) void ln_router(
    const float* __restrict__ x, const float* __restrict__ gamma,
    const float* __restrict__ beta, const float* __restrict__ Wr,
    const float* __restrict__ br, bf16_t* __restrict__ xn,
    float* __restrict__ wcomb) {
  const int token = blockIdx.x * 8 + (threadIdx.x >> 5);
  const int lane = threadIdx.x & 31;
  if (token >= NTOK) return;
  const float* xr = x + (size_t)token * DDIM;

  float v[24];
  float s = 0.f;
#pragma unroll
  for (int i = 0; i < 24; ++i) { v[i] = xr[lane + 32 * i]; s += v[i]; }
#pragma unroll
  for (int o = 16; o; o >>= 1) s += __shfl_xor(s, o, 32);
  const float mu = s * (1.0f / DDIM);

  float ss = 0.f;
#pragma unroll
  for (int i = 0; i < 24; ++i) { float d = v[i] - mu; ss += d * d; }
#pragma unroll
  for (int o = 16; o; o >>= 1) ss += __shfl_xor(ss, o, 32);
  const float rsig = rsqrtf(ss * (1.0f / DDIM) + 1e-5f);

  float r0 = 0.f, r1 = 0.f, r2 = 0.f, r3 = 0.f;
  bf16_t* xnr = xn + (size_t)token * DDIM;
  const float4* Wr4 = (const float4*)Wr;  // [D][4] rows
#pragma unroll
  for (int i = 0; i < 24; ++i) {
    const int d = lane + 32 * i;
    const float xv = (v[i] - mu) * rsig * gamma[d] + beta[d];
    xnr[d] = (bf16_t)xv;
    const float4 wr = Wr4[d];
    r0 += xv * wr.x; r1 += xv * wr.y; r2 += xv * wr.z; r3 += xv * wr.w;
  }
#pragma unroll
  for (int o = 16; o; o >>= 1) {
    r0 += __shfl_xor(r0, o, 32);
    r1 += __shfl_xor(r1, o, 32);
    r2 += __shfl_xor(r2, o, 32);
    r3 += __shfl_xor(r3, o, 32);
  }
  if (lane == 0) {
    float lg[4] = {r0 + br[0], r1 + br[1], r2 + br[2], r3 + br[3]};
    float m = lg[0];
#pragma unroll
    for (int e = 1; e < 4; ++e) m = fmaxf(m, lg[e]);
    float p[4];
#pragma unroll
    for (int e = 0; e < 4; ++e) p[e] = __expf(lg[e] - m);
    // top-2 (ties -> lowest index first, like lax.top_k)
    int i1 = 0;
#pragma unroll
    for (int e = 1; e < 4; ++e) if (p[e] > p[i1]) i1 = e;
    int i2 = (i1 == 0) ? 1 : 0;
#pragma unroll
    for (int e = 0; e < 4; ++e) if (e != i1 && e != i2 && p[e] > p[i2]) i2 = e;
    const float inv = 1.0f / (p[i1] + p[i2]);
    float4 w = {0.f, 0.f, 0.f, 0.f};
    float* wp = &w.x;
    wp[i1] = p[i1] * inv;
    wp[i2] = p[i2] * inv;
    ((float4*)wcomb)[token] = w;
  }
}

// ---------------------------------------------------------------------------
// Kernel 2: fused expert FFN + combine + residual.
// Grid: N/16 = 256 blocks x 256 threads (8 waves). Each block owns 16 tokens
// and all 768 output columns; accumulates over active experts in registers.
// xn tile staged in LDS via the Tensor Data Mover (one 3-D descriptor).
// ---------------------------------------------------------------------------
__global__ __launch_bounds__(256) void moe_ffn(
    const float* __restrict__ x, const bf16_t* __restrict__ xn,
    const float* __restrict__ wcomb,
    const bf16_t* __restrict__ W1t, const float* __restrict__ b1,
    const bf16_t* __restrict__ W2t, const float* __restrict__ b2,
    float* __restrict__ out) {
  // xn tile, chunk-major [96][16][8] bf16: TDM x->y->z fill order produces
  // exactly this; A-frag reads are 16B at (chunk*16+m)*16B -> banks 4m..4m+3.
  __shared__ __align__(16) bf16_t xn_s[96 * 16 * 8];
  __shared__ __align__(16) bf16_t h_s[16][136];  // 128 + 8 pad
  __shared__ float w_s[16][4];

  const int tile = blockIdx.x;
  const int tid = threadIdx.x;
  const int wave = tid >> 5;
  const int lane = tid & 31;
  const int ln = lane & 15;
  const int hi = lane >> 4;       // lane group: K-offset selector
  const int mbase = hi * 8;       // C rows: M = r + mbase

#if defined(USE_TDM)
  if (wave == 0) {
    // Tensor DMA: 3-D tile x=8 elems, y=16 rows (stride 768), z=96 (stride 8).
    const unsigned lds_base = (unsigned)(uintptr_t)(&xn_s[0]);
    const unsigned long long ga =
        (unsigned long long)(uintptr_t)(xn + (size_t)tile * 16 * DDIM);
    u32x4 g0 = { 1u,                                   // count=1 (valid D#)
                 lds_base,                             // lds_addr (bytes)
                 (unsigned)(ga & 0xFFFFFFFFu),         // global_addr[31:0]
                 (unsigned)((ga >> 32) & 0x01FFFFFFu)  // global_addr[56:32]
                     | (2u << 30) };                   // type=2 ("image")
    i32x8 g1 = { (int)0x00010000,        // data_size=1 (2 bytes)
                 (int)(8u  << 16),       // tensor_dim0[15:0]  = 8
                 (int)(16u << 16),       // tensor_dim1[15:0]  = 16
                 (int)(8u  << 16),       // tile_dim0 = 8
                 (int)(16u | (96u << 16)), // tile_dim1=16, tile_dim2=96
                 768,                    // tensor_dim0_stride[31:0] = 768
                 (int)(8u << 16),        // tensor_dim1_stride[15:0] = 8
                 0 };
    i32x4 g2 = { 96, 0, 0, 0 };          // tensor_dim2 = 96
    i32x4 g3 = { 0, 0, 0, 0 };
    i32x8 g4 = { 0, 0, 0, 0, 0, 0, 0, 0 };  // unused extra group (zero-filled)
    __builtin_amdgcn_tensor_load_to_lds(g0, g1, g2, g3, g4, /*cpol=*/0);
    __builtin_amdgcn_s_wait_tensorcnt(0);
  }
#else
  {
    const bf16_t* src = xn + (size_t)tile * 16 * DDIM;
#pragma unroll
    for (int c = tid; c < 16 * 96; c += 256) {
      const int r = c / 96, c8 = c - r * 96;
      *(int4v*)(xn_s + (c8 * 16 + r) * 8) = *(const int4v*)(src + r * DDIM + c8 * 8);
    }
  }
#endif
  if (tid < 64) ((float*)w_s)[tid] = wcomb[tile * 64 + tid];
  __syncthreads();

  v8f acc[6];
#pragma unroll
  for (int t = 0; t < 6; ++t) acc[t] = (v8f){0.f, 0.f, 0.f, 0.f, 0.f, 0.f, 0.f, 0.f};

  for (int e = 0; e < NEXP; ++e) {
    // Block-uniform expert skip: all threads read identical shared values.
    bool active = false;
#pragma unroll
    for (int m = 0; m < 16; ++m) active |= (w_s[m][e] != 0.0f);
    if (!active) continue;

    const bf16_t* W1e = W1t + (size_t)e * HDIM * DDIM;  // [H][D]
    const bf16_t* W2e = W2t + (size_t)e * DDIM * HDIM;  // [D][H]

    for (int kh = 0; kh < HDIM; kh += 128) {
      // ---- GEMM1: this wave computes h columns [kh + wave*16, +16), K=768
      v8f c1 = (v8f){0.f, 0.f, 0.f, 0.f, 0.f, 0.f, 0.f, 0.f};
      const int hcol = kh + wave * 16 + ln;                  // B column = N
      const bf16_t* Brow1 = W1e + (size_t)hcol * DDIM + hi * 16;
#pragma unroll 8
      for (int kd = 0; kd < DDIM; kd += 32) {
        const v16bf a = load_frag_a_xn(xn_s, ln, kd, hi);
        const v16bf b = *(const v16bf*)(Brow1 + kd);         // 32B contiguous
        c1 = wmma_bf16(a, b, c1);
      }
      // bias + ReLU + per-token router scale, store bf16 h-chunk to LDS
      const float bias = b1[e * HDIM + hcol];
#pragma unroll
      for (int r = 0; r < 8; ++r) {
        float hv = c1[r] + bias;
        hv = fmaxf(hv, 0.0f) * w_s[r + mbase][e];
        h_s[r + mbase][wave * 16 + ln] = (bf16_t)hv;
      }
      __syncthreads();

      // ---- GEMM2: this wave owns d columns [wave*96, +96) (6 tiles), K=128
#pragma unroll
      for (int t = 0; t < 6; ++t) {
        const int dcol = wave * 96 + t * 16 + ln;
        const bf16_t* Brow2 = W2e + (size_t)dcol * HDIM + kh + hi * 16;
        if (kh + 128 < HDIM) __builtin_prefetch(Brow2 + 128, 0, 0);
#pragma unroll
        for (int kk = 0; kk < 128; kk += 32) {
          const v16bf a = load_frag_a_row(&h_s[ln][0], kk, hi);
          const v16bf b = *(const v16bf*)(Brow2 + kk);       // 32B contiguous
          acc[t] = wmma_bf16(a, b, acc[t]);
        }
      }
      __syncthreads();  // protect h_s before next chunk's stores
    }
  }

  // ---- Epilogue: out = x + sum_e w*(h@W2) + sum_e w*b2
#pragma unroll
  for (int t = 0; t < 6; ++t) {
    const int dcol = wave * 96 + t * 16 + ln;
#pragma unroll
    for (int r = 0; r < 8; ++r) {
      const int row = r + mbase;
      const int token = tile * 16 + row;
      float bsum = 0.f;
#pragma unroll
      for (int e = 0; e < NEXP; ++e) bsum += w_s[row][e] * b2[e * DDIM + dcol];
      out[(size_t)token * DDIM + dcol] =
          x[(size_t)token * DDIM + dcol] + acc[t][r] + bsum;
    }
  }
}

// ---------------------------------------------------------------------------
// Launch: prep (weights->bf16 transposed), ln+router, fused FFN.
// Workspace layout: xn_bf16 | wcomb f32 | W1t bf16 | W2t bf16  (~44.1 MB)
// ---------------------------------------------------------------------------
extern "C" void kernel_launch(void* const* d_in, const int* in_sizes, int n_in,
                              void* d_out, int out_size, void* d_ws, size_t ws_size,
                              hipStream_t stream) {
  const float* x     = (const float*)d_in[0];
  const float* gamma = (const float*)d_in[1];
  const float* beta  = (const float*)d_in[2];
  const float* Wr    = (const float*)d_in[3];
  const float* br    = (const float*)d_in[4];
  const float* W1    = (const float*)d_in[5];
  const float* b1    = (const float*)d_in[6];
  const float* W2    = (const float*)d_in[7];
  const float* b2    = (const float*)d_in[8];
  float* out = (float*)d_out;

  char* ws = (char*)d_ws;
  size_t off = 0;
  bf16_t* xn = (bf16_t*)(ws + off);
  off += (size_t)NTOK * DDIM * sizeof(bf16_t);           // 6.29 MB
  float* wcomb = (float*)(ws + off);
  off += (size_t)NTOK * NEXP * sizeof(float);            // 64 KB
  off = (off + 255) & ~(size_t)255;
  bf16_t* W1t = (bf16_t*)(ws + off);
  off += (size_t)NEXP * HDIM * DDIM * sizeof(bf16_t);    // 18.87 MB
  bf16_t* W2t = (bf16_t*)(ws + off);
  off += (size_t)NEXP * HDIM * DDIM * sizeof(bf16_t);    // 18.87 MB

  prep_weights<<<4096, 256, 0, stream>>>(W1, W2, W1t, W2t);
  ln_router<<<NTOK / 8, 256, 0, stream>>>(x, gamma, beta, Wr, br, xn, wcomb);
  moe_ffn<<<NTOK / 16, 256, 0, stream>>>(x, xn, wcomb, W1t, b1, W2t, b2, out);
}
